// DynamicVFE_31834297598160
// MI455X (gfx1250) — matrix operand order repaired and checked by reading, not compiled, not run on hardware
//
#include <hip/hip_runtime.h>
#include <hip/hip_bf16.h>

typedef __attribute__((ext_vector_type(16))) _Float16 v16h;
typedef __attribute__((ext_vector_type(8)))  float    v8f;

#define BN_EPS 1e-3f
#define VXc 0.2f
#define VYc 0.2f
#define VZc 4.0f
#define XOFF 0.1f
#define YOFF (-39.9f)
#define ZOFF (-1.0f)

// ---------------------------------------------------------------------------
// 1) init: zero the max/sum accumulators + output feature region (identity for
//    atomicMax on post-ReLU values is 0), and emit voxel_coors as floats.
// ---------------------------------------------------------------------------
__global__ void __launch_bounds__(256)
vfe_init(float* __restrict__ out_feats, float* __restrict__ v0f,
         float* __restrict__ vsum, float* __restrict__ vcnt,
         const int* __restrict__ vcoors, float* __restrict__ out_coors, int M)
{
    long i = (long)blockIdx.x * 256 + threadIdx.x;
    long tot = (long)M * 128;
    if (i < tot)          out_feats[i] = 0.0f;
    if (i < (long)M * 64) v0f[i] = 0.0f;
    if (i < (long)M * 4)  { vsum[i] = 0.0f; out_coors[i] = (float)vcoors[i]; }
    if (i < (long)M)      vcnt[i] = 0.0f;
}

// ---------------------------------------------------------------------------
// 2) fold BN (eval mode) into the GEMM weights:
//    relu(bn(x@W)) == relu(x@(W*s) + (beta - mean*s)),  s = gamma*rsqrt(var+eps)
//    W0 stays f32 (tiny VALU GEMM). W1 is folded to f16 AND pre-swizzled into
//    WMMA B-fragment order so layer 1 fetches fragments with b128 loads:
//      index = ((strip*4 + chunk)*32 + lane)*16 + e
//      K     = chunk*32 + 16*(lane>>4) + e,  col = strip*16 + (lane&15)
// ---------------------------------------------------------------------------
__global__ void __launch_bounds__(256)
vfe_fold(const float* __restrict__ W0, const float* __restrict__ g0,
         const float* __restrict__ be0, const float* __restrict__ m0,
         const float* __restrict__ va0,
         const float* __restrict__ W1, const float* __restrict__ g1,
         const float* __restrict__ be1, const float* __restrict__ m1,
         const float* __restrict__ va1,
         float* __restrict__ W0s, float* __restrict__ b0f,
         _Float16* __restrict__ W1frag, float* __restrict__ b1f)
{
    __shared__ float s1[128];
    int t = threadIdx.x;
    if (t < 64) {
        float s = g0[t] * rsqrtf(va0[t] + BN_EPS);
        b0f[t] = be0[t] - m0[t] * s;
        for (int j = 0; j < 11; ++j) W0s[j * 64 + t] = W0[j * 64 + t] * s;
    }
    if (t < 128) {
        float s = g1[t] * rsqrtf(va1[t] + BN_EPS);
        s1[t] = s;
        b1f[t] = be1[t] - m1[t] * s;
    }
    __syncthreads();
    for (int i = t; i < 128 * 128; i += 256) {
        int e    = i & 15;
        int lane = (i >> 4) & 31;
        int c    = (i >> 9) & 3;
        int cb   = i >> 11;
        int k    = c * 32 + ((lane >> 4) << 4) + e;
        int col  = cb * 16 + (lane & 15);
        W1frag[i] = (_Float16)(W1[k * 128 + col] * s1[col]);
    }
}

// ---------------------------------------------------------------------------
// 3) segment-sum of raw features (for per-voxel mean) via f32 global atomics.
// ---------------------------------------------------------------------------
__global__ void __launch_bounds__(256)
vfe_mean_scatter(const float* __restrict__ feats, const int* __restrict__ seg_ids,
                 float* __restrict__ vsum, float* __restrict__ vcnt, int N)
{
    int p = blockIdx.x * 256 + threadIdx.x;
    if (p >= N) return;
    int seg = seg_ids[p];
    atomicAdd(&vsum[(long)seg * 4 + 0], feats[(long)p * 4 + 0]);
    atomicAdd(&vsum[(long)seg * 4 + 1], feats[(long)p * 4 + 1]);
    atomicAdd(&vsum[(long)seg * 4 + 2], feats[(long)p * 4 + 2]);
    atomicAdd(&vsum[(long)seg * 4 + 3], feats[(long)p * 4 + 3]);
    atomicAdd(&vcnt[seg], 1.0f);
}

// ---------------------------------------------------------------------------
// 4) layer 0: build 11-d feature, x@W0s + b0, ReLU; h0 -> f16 (WMMA feed),
//    segment-max into v0 (uint atomicMax is exact for non-negative floats).
// ---------------------------------------------------------------------------
__global__ void __launch_bounds__(256)
vfe_layer0(const float* __restrict__ feats, const int* __restrict__ coors,
           const int* __restrict__ seg_ids,
           const float* __restrict__ vsum, const float* __restrict__ vcnt,
           const float* __restrict__ W0s, const float* __restrict__ b0f,
           _Float16* __restrict__ h0h, unsigned int* __restrict__ v0u, int N)
{
    __shared__ float sW[11 * 64];
    __shared__ float sB[64];
    for (int i = threadIdx.x; i < 11 * 64; i += 256) sW[i] = W0s[i];
    if (threadIdx.x < 64) sB[threadIdx.x] = b0f[threadIdx.x];
    __syncthreads();

    int p = blockIdx.x * 256 + threadIdx.x;
    if (p >= N) return;

    float fx = feats[(long)p * 4 + 0], fy = feats[(long)p * 4 + 1];
    float fz = feats[(long)p * 4 + 2], fi = feats[(long)p * 4 + 3];
    int seg = seg_ids[p];
    float inv = 1.0f / fmaxf(vcnt[seg], 1.0f);
    float mx = vsum[(long)seg * 4 + 0] * inv;
    float my = vsum[(long)seg * 4 + 1] * inv;
    float mz = vsum[(long)seg * 4 + 2] * inv;
    int cx = coors[(long)p * 4 + 3], cy = coors[(long)p * 4 + 2], cz = coors[(long)p * 4 + 1];

    float x[11];
    x[0] = fx; x[1] = fy; x[2] = fz; x[3] = fi;
    x[4] = fx - mx; x[5] = fy - my; x[6] = fz - mz;
    x[7] = fx - ((float)cx * VXc + XOFF);
    x[8] = fy - ((float)cy * VYc + YOFF);
    x[9] = fz - ((float)cz * VZc + ZOFF);
    x[10] = sqrtf(fx * fx + fy * fy + fz * fz);

    unsigned int segb = (unsigned int)seg * 64u;
#pragma unroll 8
    for (int c = 0; c < 64; ++c) {
        float a = sB[c];
#pragma unroll
        for (int j = 0; j < 11; ++j) a = fmaf(x[j], sW[j * 64 + c], a);
        a = fmaxf(a, 0.0f);
        h0h[(long)p * 64 + c] = (_Float16)a;
        atomicMax(&v0u[segb + c], __float_as_uint(a));
    }
}

// ---------------------------------------------------------------------------
// 5) layer 1: WMMA GEMM  x1[128pts,128] @ W1[128,128]  (f16 in, f32 acc),
//    bias + ReLU epilogue, segment-max scatter to output voxels.
//    Fragment layouts used here (wave32):
//      A: lane -> row M=lane&15;   elem e -> K = 16*(lane>>4) + e
//      B: lane -> col N=lane&15;   elem e -> K = 16*(lane>>4) + e   (pre-swizzled)
//      D: lane -> col N=lane&15;   vgpr r -> row M = r + 8*(lane>>4)
//    Scatter uses 32-bit element indices off the scalar out-base (GVS mode),
//    avoiding per-element 64-bit address VALU chains.
// ---------------------------------------------------------------------------
#define X1_STRIDE 136  // 128 + 8 halves: shifts LDS banks by 4 per row -> no conflicts

template <bool FULL>
__device__ __forceinline__ void layer1_compute(
    const _Float16* __restrict__ sX, const unsigned int* __restrict__ sSegOff,
    const v16h* __restrict__ bfr, float bias,
    unsigned int* __restrict__ outu, int pbase, int N,
    int ln15, int lh, unsigned int outcol)
{
#pragma unroll
    for (int rt = 0; rt < 8; ++rt) {
        v8f acc = {};
#pragma unroll
        for (int c = 0; c < 4; ++c) {
            v16h af;
            const _Float16* arow = &sX[(rt * 16 + ln15) * X1_STRIDE + c * 32 + lh * 16];
            *(uint4*)&(((_Float16*)&af)[0]) = *(const uint4*)&arow[0];   // K = base+16lh .. +7
            *(uint4*)&(((_Float16*)&af)[8]) = *(const uint4*)&arow[8];   // K = base+16lh+8 ..
            acc = __builtin_amdgcn_wmma_f32_16x16x32_f16(
                false, af, false, bfr[c], (short)0, acc, false, false);
        }
#pragma unroll
        for (int r = 0; r < 8; ++r) {
            int row = rt * 16 + r + lh * 8;
            if (FULL || (pbase + row < N)) {
                float v = fmaxf(acc[r] + bias, 0.0f);
                atomicMax(&outu[sSegOff[row] + outcol], __float_as_uint(v));
            }
        }
    }
}

__global__ void __launch_bounds__(256)
vfe_layer1_wmma(const _Float16* __restrict__ h0h, const float* __restrict__ v0f,
                const _Float16* __restrict__ W1frag, const float* __restrict__ b1f,
                const int* __restrict__ seg_ids, float* __restrict__ out, int N)
{
    __shared__ _Float16 sX[128 * X1_STRIDE];
    __shared__ unsigned int sSegOff[128];   // seg*128, pre-scaled element offset

    const int t = threadIdx.x;
    const int pbase = blockIdx.x * 128;

    // prefetch next block's h0 tile (16KB) into cache while we compute
    if (pbase + 128 < N)
        __builtin_prefetch(&h0h[(long)(pbase + 128) * 64 + t * 32], 0, 1);

    if (t < 128) {
        int p = pbase + t;
        sSegOff[t] = (p < N) ? ((unsigned int)seg_ids[p] << 7) : 0u;
    }
    // stage x1 = [h0 | v0[seg]] as f16: thread t -> row t>>1, half t&1 (64 halves)
    {
        int row = t >> 1, half = t & 1;
        int p = pbase + row;
        _Float16* dst = &sX[row * X1_STRIDE + half * 64];
        if (p < N) {
            if (half == 0) {
                const uint4* src = (const uint4*)&h0h[(long)p * 64];
                uint4* d4 = (uint4*)dst;
#pragma unroll
                for (int i = 0; i < 8; ++i) d4[i] = src[i];
            } else {
                const float4* src = (const float4*)&v0f[(long)seg_ids[p] * 64];
#pragma unroll
                for (int i = 0; i < 16; ++i) {
                    float4 f = src[i];
                    dst[i * 4 + 0] = (_Float16)f.x;
                    dst[i * 4 + 1] = (_Float16)f.y;
                    dst[i * 4 + 2] = (_Float16)f.z;
                    dst[i * 4 + 3] = (_Float16)f.w;
                }
            }
        } else {
            uint4 z = {0u, 0u, 0u, 0u};
            uint4* d4 = (uint4*)dst;
#pragma unroll
            for (int i = 0; i < 8; ++i) d4[i] = z;
        }
    }
    __syncthreads();

    const int lane = t & 31;
    const int wv   = t >> 5;        // 8 waves -> 8 column strips of 16
    const int colb = wv * 16;
    const int ln15 = lane & 15;
    const int lh   = lane >> 4;

    // B fragments for all four K chunks: pre-swizzled -> 2x b128 loads per chunk
    v16h bfr[4];
#pragma unroll
    for (int c = 0; c < 4; ++c) {
        const _Float16* bp = &W1frag[(((wv * 4 + c) * 32 + lane) << 4)];
        *(uint4*)&(((_Float16*)&bfr[c])[0]) = *(const uint4*)&bp[0];
        *(uint4*)&(((_Float16*)&bfr[c])[8]) = *(const uint4*)&bp[8];
    }
    const float bias = b1f[colb + ln15];
    const unsigned int outcol = (unsigned int)(colb + ln15);

    if (pbase + 128 <= N)
        layer1_compute<true >(sX, sSegOff, bfr, bias, (unsigned int*)out,
                              pbase, N, ln15, lh, outcol);
    else
        layer1_compute<false>(sX, sSegOff, bfr, bias, (unsigned int*)out,
                              pbase, N, ln15, lh, outcol);
}

// ---------------------------------------------------------------------------
extern "C" void kernel_launch(void* const* d_in, const int* in_sizes, int n_in,
                              void* d_out, int out_size, void* d_ws, size_t ws_size,
                              hipStream_t stream)
{
    const float* feats  = (const float*)d_in[0];
    const float* W0     = (const float*)d_in[1];
    const float* g0     = (const float*)d_in[2];
    const float* be0    = (const float*)d_in[3];
    const float* m0     = (const float*)d_in[4];
    const float* va0    = (const float*)d_in[5];
    const float* W1     = (const float*)d_in[6];
    const float* g1     = (const float*)d_in[7];
    const float* be1    = (const float*)d_in[8];
    const float* m1     = (const float*)d_in[9];
    const float* va1    = (const float*)d_in[10];
    const int*   coors  = (const int*)d_in[11];
    const int*   seg    = (const int*)d_in[12];
    const int*   vcoors = (const int*)d_in[13];

    const int N = in_sizes[0] / 4;
    const int M = in_sizes[13] / 4;

    char* ws = (char*)d_ws;
    size_t off = 0;
    auto carve = [&](size_t bytes) {
        void* p = ws + off;
        off += (bytes + 255) & ~(size_t)255;
        return p;
    };
    _Float16* h0h    = (_Float16*)carve((size_t)N * 64 * 2);
    float*    v0f    = (float*)carve((size_t)M * 64 * 4);
    float*    vsum   = (float*)carve((size_t)M * 4 * 4);
    float*    vcnt   = (float*)carve((size_t)M * 4);
    float*    W0s    = (float*)carve(11 * 64 * 4);
    float*    b0f    = (float*)carve(64 * 4);
    _Float16* W1frag = (_Float16*)carve(128 * 128 * 2);
    float*    b1f    = (float*)carve(128 * 4);

    float* out_feats = (float*)d_out;               // [M,128]
    float* out_coors = out_feats + (size_t)M * 128; // [M,4] as float

    long initTot = (long)M * 128;
    int  initBlocks = (int)((initTot + 255) / 256);
    vfe_init<<<initBlocks, 256, 0, stream>>>(out_feats, v0f, vsum, vcnt, vcoors, out_coors, M);

    vfe_fold<<<1, 256, 0, stream>>>(W0, g0, be0, m0, va0,
                                    W1, g1, be1, m1, va1,
                                    W0s, b0f, W1frag, b1f);

    vfe_mean_scatter<<<(N + 255) / 256, 256, 0, stream>>>(feats, seg, vsum, vcnt, N);

    vfe_layer0<<<(N + 255) / 256, 256, 0, stream>>>(feats, coors, seg, vsum, vcnt,
                                                    W0s, b0f, h0h, (unsigned int*)v0f, N);

    vfe_layer1_wmma<<<(N + 127) / 128, 256, 0, stream>>>(h0h, v0f, W1frag, b1f, seg,
                                                         out_feats, N);
}
